// TTTLanguageModel_37374805409899
// MI455X (gfx1250) — matrix operand order repaired and checked by reading, not compiled, not run on hardware
//
#include <hip/hip_runtime.h>
#include <cstdint>
#include <cstddef>

// ---------------------------------------------------------------------------
// Types for CDNA5 WMMA (wave32)
// ---------------------------------------------------------------------------
typedef __attribute__((ext_vector_type(8)))  __bf16 v8bf;
typedef __attribute__((ext_vector_type(16))) __bf16 v16bf;
typedef __attribute__((ext_vector_type(8)))  float  v8f;

// Model constants
#define Dm   768
#define Hh   12
#define DHd  64
#define Ll   12
#define Vv   50257
#define Bb   4
#define Ss   1024
#define Tt   (Bb * Ss)   // 4096
#define Mm   16
#define NC   64          // chunks per sequence
#define ETA  (1.0f / 16.0f)

// ---------------------------------------------------------------------------
// CDNA5 async global->LDS copy (ASYNCcnt-tracked, bypasses VGPRs)
// ---------------------------------------------------------------------------
__device__ __forceinline__ void async_copy_b128(const void* gsrc,
                                                const void* lds_dst)
{
    unsigned           loff  = (unsigned)(size_t)lds_dst;   // low 32b = LDS off
    unsigned long long gaddr = (unsigned long long)(size_t)gsrc;
    asm volatile("global_load_async_to_lds_b128 %0, %1, off"
                 :: "v"(loff), "v"(gaddr) : "memory");
}
__device__ __forceinline__ void wait_async0()
{
    asm volatile("s_wait_asynccnt 0" ::: "memory");
}

__device__ __forceinline__ float gelu_tanh(float x)
{
    float t = tanhf(0.7978845608028654f * (x + 0.044715f * x * x * x));
    return 0.5f * x * (1.0f + t);
}

// ---------------------------------------------------------------------------
// Weight convert + transpose: W (K x N fp32) -> Wt (N x K bf16),
// separate per-layer strides so QKV can pack into one N=2304 weight.
// grid: (N/32, K/32, L), 256 threads
// ---------------------------------------------------------------------------
__global__ __launch_bounds__(256) void conv_transpose_kernel(
    const float* __restrict__ W, __bf16* __restrict__ Wt, int K, int N,
    size_t in_ls, size_t out_ls)
{
    __shared__ float tile[32][33];
    const int tx = threadIdx.x & 31;
    const int ty = threadIdx.x >> 5;            // 0..7
    const int n0 = blockIdx.x * 32;
    const int k0 = blockIdx.y * 32;
    const float* Wl  = W  + (size_t)blockIdx.z * in_ls;
    __bf16*      Wtl = Wt + (size_t)blockIdx.z * out_ls;
#pragma unroll
    for (int i = 0; i < 4; ++i) {
        int k = ty + i * 8;
        tile[k][tx] = Wl[(size_t)(k0 + k) * N + n0 + tx];
    }
    __syncthreads();
#pragma unroll
    for (int i = 0; i < 4; ++i) {
        int n = ty + i * 8;
        Wtl[(size_t)(n0 + n) * K + k0 + tx] = (__bf16)tile[tx][n];
    }
}

// ---------------------------------------------------------------------------
// Plain fp32 -> bf16 cast (tok_emb is already N x K for the logits GEMM)
// ---------------------------------------------------------------------------
__global__ __launch_bounds__(256) void cast_bf16_kernel(
    const float* __restrict__ in, __bf16* __restrict__ out, size_t n)
{
    size_t i = (size_t)blockIdx.x * 256 + threadIdx.x;
    if (i < n) out[i] = (__bf16)in[i];
}

// ---------------------------------------------------------------------------
// Embedding gather
// ---------------------------------------------------------------------------
__global__ __launch_bounds__(256) void gather_embed_kernel(
    const int* __restrict__ ids, const float* __restrict__ emb,
    float* __restrict__ x)
{
    int t  = blockIdx.x;
    int id = ids[t];
    for (int i = threadIdx.x; i < Dm; i += 256)
        x[(size_t)t * Dm + i] = emb[(size_t)id * Dm + i];
}

// ---------------------------------------------------------------------------
// LayerNorm over D=768 (3 elems/thread), OutT = float or __bf16
// ---------------------------------------------------------------------------
template <typename OutT>
__global__ __launch_bounds__(256) void ln_kernel(
    const float* __restrict__ x, const float* __restrict__ w,
    const float* __restrict__ b, OutT* __restrict__ out)
{
    const int row = blockIdx.x;
    const float* xr = x + (size_t)row * Dm;
    float vals[3];
    float lsum = 0.f, lsq = 0.f;
#pragma unroll
    for (int i = 0; i < 3; ++i) {
        float v = xr[threadIdx.x + i * 256];
        vals[i] = v; lsum += v; lsq += v * v;
    }
    __shared__ float s1[256], s2[256];
    s1[threadIdx.x] = lsum; s2[threadIdx.x] = lsq;
    __syncthreads();
    for (int off = 128; off > 0; off >>= 1) {
        if (threadIdx.x < off) {
            s1[threadIdx.x] += s1[threadIdx.x + off];
            s2[threadIdx.x] += s2[threadIdx.x + off];
        }
        __syncthreads();
    }
    const float mean = s1[0] * (1.0f / Dm);
    const float var  = s2[0] * (1.0f / Dm) - mean * mean;
    const float rstd = rsqrtf(var + 1e-5f);
#pragma unroll
    for (int i = 0; i < 3; ++i) {
        int c = threadIdx.x + i * 256;
        out[(size_t)row * Dm + c] = (OutT)((vals[i] - mean) * rstd * w[c] + b[c]);
    }
}

// ---------------------------------------------------------------------------
// RoPE in place on q,k inside the packed qkv (T x 3D) buffer
// ---------------------------------------------------------------------------
__global__ __launch_bounds__(256) void rope_kernel(float* __restrict__ qkv)
{
    int idx = blockIdx.x * 256 + threadIdx.x;
    if (idx >= Tt * Hh * 32) return;
    const int i = idx & 31;
    const int h = (idx >> 5) % Hh;
    const int t = idx / (32 * Hh);
    const int s = t & (Ss - 1);
    const float freq = __powf(10000.0f, -(float)i * (1.0f / 32.0f));
    float sn, cs;
    __sincosf((float)s * freq, &sn, &cs);
    const size_t base = (size_t)t * (3 * Dm) + h * DHd;
    float x1 = qkv[base + i], x2 = qkv[base + 32 + i];
    qkv[base + i]      = x1 * cs - x2 * sn;
    qkv[base + 32 + i] = x1 * sn + x2 * cs;
    x1 = qkv[base + Dm + i]; x2 = qkv[base + Dm + 32 + i];
    qkv[base + Dm + i]      = x1 * cs - x2 * sn;
    qkv[base + Dm + 32 + i] = x1 * sn + x2 * cs;
}

// ---------------------------------------------------------------------------
// TTT scan: one block per (b,h), sequential over 64 chunks of M=16.
// q/k/v tiles are double-buffered in LDS via async copies so the next
// chunk's 12 KB stream overlaps with the current chunk's compute.
// Reads packed qkv (T x 3D fp32), writes O as bf16 (T x D).
// ---------------------------------------------------------------------------
__global__ __launch_bounds__(256) void ttt_scan_kernel(
    const float* __restrict__ qkv, __bf16* __restrict__ o)
{
    const int b = blockIdx.x / Hh;
    const int h = blockIdx.x % Hh;
    __shared__ float Wst[64][65];
    __shared__ float Qm[2][16][64], Km[2][16][64], Vm[2][16][64];
    __shared__ float Em[16][65];
    __shared__ float Am[16][16];
    const int tid = threadIdx.x;

    for (int i = tid; i < 64 * 64; i += 256) Wst[i >> 6][i & 63] = 0.f;

    const int m  = tid >> 4;   // 0..15
    const int dl = tid & 15;   // 0..15 (d = dl + 16*j)

    // async chunk prefetch: thread -> one b128 (4 floats) per matrix
    const int pr = tid >> 4;          // row 0..15
    const int pd = (tid & 15) * 4;    // col 0,4,...,60
    auto issue_chunk = [&](int buf, int c) {
        size_t g = (size_t)(b * Ss + c * Mm + pr) * (3 * Dm) + h * DHd + pd;
        async_copy_b128(qkv + g,          &Qm[buf][pr][pd]);
        async_copy_b128(qkv + g + Dm,     &Km[buf][pr][pd]);
        async_copy_b128(qkv + g + 2 * Dm, &Vm[buf][pr][pd]);
    };

    issue_chunk(0, 0);

    for (int c = 0; c < NC; ++c) {
        const int buf = c & 1;
        wait_async0();                 // this chunk's q/k/v landed in LDS
        __syncthreads();               // (also orders W zero-init / W update)
        if (c + 1 < NC) issue_chunk(buf ^ 1, c + 1);   // overlap next stream

        // E = Km @ W - Vm ; A = (Qm Km^T) * tril
        {
            float acc[4] = {0.f, 0.f, 0.f, 0.f};
            for (int e = 0; e < 64; ++e) {
                float kv = Km[buf][m][e];
#pragma unroll
                for (int j = 0; j < 4; ++j) acc[j] += kv * Wst[e][dl + j * 16];
            }
#pragma unroll
            for (int j = 0; j < 4; ++j)
                Em[m][dl + j * 16] = acc[j] - Vm[buf][m][dl + j * 16];
            float a = 0.f;
            for (int d = 0; d < 64; ++d) a += Qm[buf][m][d] * Km[buf][dl][d];
            Am[m][dl] = (dl <= m) ? a : 0.f;
        }
        __syncthreads();

        // O = Qm @ W - eta * (A @ E) -> bf16
        {
            float o1[4] = {0.f, 0.f, 0.f, 0.f};
            for (int e = 0; e < 64; ++e) {
                float qv = Qm[buf][m][e];
#pragma unroll
                for (int j = 0; j < 4; ++j) o1[j] += qv * Wst[e][dl + j * 16];
            }
            float o2[4] = {0.f, 0.f, 0.f, 0.f};
#pragma unroll
            for (int n = 0; n < 16; ++n) {
                float av = Am[m][n];
#pragma unroll
                for (int j = 0; j < 4; ++j) o2[j] += av * Em[n][dl + j * 16];
            }
            size_t gb = (size_t)(b * Ss + c * Mm + m) * Dm + h * DHd;
#pragma unroll
            for (int j = 0; j < 4; ++j)
                o[gb + dl + j * 16] = (__bf16)(o1[j] - ETA * o2[j]);
        }
        __syncthreads();

        // W -= eta * Km^T @ E
        for (int i = tid; i < 4096; i += 256) {
            int e = i >> 6, d = i & 63;
            float s = 0.f;
#pragma unroll
            for (int mm = 0; mm < 16; ++mm) s += Km[buf][mm][e] * Em[mm][d];
            Wst[e][d] -= ETA * s;
        }
        __syncthreads();
    }
}

// ---------------------------------------------------------------------------
// WMMA GEMM, async double-buffered:
//   C(MxN fp32) = A(MxK bf16) @ Bt(NxK bf16)^T [+bias] [+res]
//   or Cb(MxN bf16) = gelu(A@B + bias)           (Cb != null)
// Block 128x128x32, 8 waves x (2x4) wmma tiles. M,K multiples of 128/32.
// N-edge handled by clamping B rows (garbage cols never stored).
// ---------------------------------------------------------------------------
__global__ __launch_bounds__(256) void gemm_bf16_kernel(
    const __bf16* __restrict__ A, const __bf16* __restrict__ Bt,
    float* __restrict__ C, __bf16* __restrict__ Cb,
    const float* __restrict__ bias, const float* __restrict__ res,
    int Mr, int Nr, int Kr)
{
    __shared__ __bf16 As[2][128][40];   // 32 K + 8 pad, conflict-free
    __shared__ __bf16 Bs[2][128][40];

    const int tid  = threadIdx.x;
    const int lane = tid & 31;
    const int wave = tid >> 5;
    const int wm   = wave & 3;       // 4 waves along M
    const int wn   = wave >> 2;      // 2 waves along N
    const int bm0  = blockIdx.y * 128;
    const int bn0  = blockIdx.x * 128;
    const int lrow = lane & 15;
    const int lhi  = lane >> 4;

    // per-thread staging chunks: 2 x 16B for A, 2 x 16B for B per K-step
    const int r0 = tid >> 2;                 // 0..63
    const int r1 = r0 + 64;                  // 64..127
    const int kc = (tid & 3) * 8;
    const int bnr0 = min(bn0 + r0, Nr - 1);  // clamp N edge
    const int bnr1 = min(bn0 + r1, Nr - 1);
    const __bf16* Ab0 = A + (size_t)(bm0 + r0) * Kr + kc;
    const __bf16* Ab1 = A + (size_t)(bm0 + r1) * Kr + kc;
    const __bf16* Bb0 = Bt + (size_t)bnr0 * Kr + kc;
    const __bf16* Bb1 = Bt + (size_t)bnr1 * Kr + kc;

    auto issue = [&](int buf, int k0) {
        async_copy_b128(Ab0 + k0, &As[buf][r0][kc]);
        async_copy_b128(Ab1 + k0, &As[buf][r1][kc]);
        async_copy_b128(Bb0 + k0, &Bs[buf][r0][kc]);
        async_copy_b128(Bb1 + k0, &Bs[buf][r1][kc]);
    };

    v8f acc[2][4];
#pragma unroll
    for (int t = 0; t < 2; ++t)
#pragma unroll
        for (int u = 0; u < 4; ++u)
#pragma unroll
            for (int r = 0; r < 8; ++r) acc[t][u][r] = 0.0f;

    issue(0, 0);                       // prologue: fill buffer 0
    int buf = 0;
    for (int k0 = 0; k0 < Kr; k0 += 32) {
        wait_async0();                 // current buffer landed in LDS
        __syncthreads();               // visible to all waves; prev reads done
        if (k0 + 32 < Kr) issue(buf ^ 1, k0 + 32);   // overlap with compute

        // A frag: lane L -> M=L%16; K = (L/16)*8 + e (e<8), 16+(L/16)*8+(e-8)
        v16bf afrag[2];
#pragma unroll
        for (int t = 0; t < 2; ++t) {
            int row = wm * 32 + t * 16 + lrow;
            int klo = lhi * 8;
            v8bf lo = *(const v8bf*)(&As[buf][row][klo]);
            v8bf hi = *(const v8bf*)(&As[buf][row][klo + 16]);
            afrag[t] = __builtin_shufflevector(lo, hi, 0, 1, 2, 3, 4, 5, 6, 7,
                                               8, 9, 10, 11, 12, 13, 14, 15);
        }
        // B frag: lane L -> N=L%16; K = (L/16)*8 + e (e<8), 16+(L/16)*8+(e-8)
        v16bf bfrag[4];
#pragma unroll
        for (int u = 0; u < 4; ++u) {
            int col = wn * 64 + u * 16 + lrow;
            int klo = lhi * 8;
            v8bf lo = *(const v8bf*)(&Bs[buf][col][klo]);
            v8bf hi = *(const v8bf*)(&Bs[buf][col][klo + 16]);
            bfrag[u] = __builtin_shufflevector(lo, hi, 0, 1, 2, 3, 4, 5, 6, 7,
                                               8, 9, 10, 11, 12, 13, 14, 15);
        }
#pragma unroll
        for (int t = 0; t < 2; ++t)
#pragma unroll
            for (int u = 0; u < 4; ++u)
                acc[t][u] = __builtin_amdgcn_wmma_f32_16x16x32_bf16(
                    false, afrag[t], false, bfrag[u], (short)0, acc[t][u],
                    false, false);
        buf ^= 1;
    }

    // epilogue: VGPR r -> M = tile_m + lhi*8 + r, N = lrow
#pragma unroll
    for (int t = 0; t < 2; ++t) {
        int row = bm0 + wm * 32 + t * 16 + lhi * 8;
#pragma unroll
        for (int u = 0; u < 4; ++u) {
            int col = bn0 + wn * 64 + u * 16 + lrow;
            if (col < Nr) {
                float bv = bias ? bias[col] : 0.0f;
#pragma unroll
                for (int r = 0; r < 8; ++r) {
                    float  val = acc[t][u][r] + bv;
                    size_t idx = (size_t)(row + r) * Nr + col;
                    if (Cb) {
                        Cb[idx] = (__bf16)gelu_tanh(val);
                    } else {
                        if (res) val += res[idx];
                        C[idx] = val;
                    }
                }
            }
        }
    }
}

// ---------------------------------------------------------------------------
// Host orchestration
// ---------------------------------------------------------------------------
extern "C" void kernel_launch(void* const* d_in, const int* in_sizes, int n_in,
                              void* d_out, int out_size, void* d_ws,
                              size_t ws_size, hipStream_t stream)
{
    const int*   input_ids = (const int*)d_in[0];
    const float* tok_emb   = (const float*)d_in[1];
    const float* emb_ln_w  = (const float*)d_in[2];
    const float* emb_ln_b  = (const float*)d_in[3];
    const float* wq        = (const float*)d_in[4];
    const float* wk        = (const float*)d_in[5];
    const float* wv        = (const float*)d_in[6];
    const float* wo        = (const float*)d_in[7];
    const float* n1w       = (const float*)d_in[8];
    const float* n1b       = (const float*)d_in[9];
    const float* n2w       = (const float*)d_in[10];
    const float* n2b       = (const float*)d_in[11];
    const float* m1w       = (const float*)d_in[12];
    const float* m1b       = (const float*)d_in[13];
    const float* m2w       = (const float*)d_in[14];
    const float* m2b       = (const float*)d_in[15];
    const float* onw       = (const float*)d_in[16];
    const float* onb       = (const float*)d_in[17];

    char* ws = (char*)d_ws;
    size_t off = 0;
    auto alloc = [&](size_t bytes) -> void* {
        void* p = ws + off;
        off = (off + bytes + 255) & ~(size_t)255;
        return p;
    };

    const size_t nDD = (size_t)Ll * Dm * Dm;
    const size_t nD4 = (size_t)Ll * Dm * 4 * Dm;

    __bf16* wqkv_t = (__bf16*)alloc(3 * nDD * 2);   // [L][3D][D] packed N x K
    __bf16* wo_t   = (__bf16*)alloc(nDD * 2);
    __bf16* m1w_t  = (__bf16*)alloc(nD4 * 2);
    __bf16* m2w_t  = (__bf16*)alloc(nD4 * 2);
    __bf16* embbf  = (__bf16*)alloc((size_t)Vv * Dm * 2);
    float*  x      = (float*)alloc((size_t)Tt * Dm * 4);
    __bf16* h_bf   = (__bf16*)alloc((size_t)Tt * Dm * 2);
    float*  qkv    = (float*)alloc((size_t)Tt * 3 * Dm * 4);
    __bf16* o_bf   = (__bf16*)alloc((size_t)Tt * Dm * 2);
    __bf16* g_bf   = (__bf16*)alloc((size_t)Tt * 4 * Dm * 2);
    (void)ws_size; (void)n_in; (void)in_sizes; (void)out_size;

    // ---- weight prep (bf16, transposed to N x K; QKV packed) ----
    const size_t sDD = (size_t)Dm * Dm;
    const size_t sD4 = (size_t)Dm * 4 * Dm;
    dim3 tDD(Dm / 32, Dm / 32, Ll);
    conv_transpose_kernel<<<tDD, 256, 0, stream>>>(wq, wqkv_t,          Dm, Dm,
                                                   sDD, 3 * sDD);
    conv_transpose_kernel<<<tDD, 256, 0, stream>>>(wk, wqkv_t + sDD,    Dm, Dm,
                                                   sDD, 3 * sDD);
    conv_transpose_kernel<<<tDD, 256, 0, stream>>>(wv, wqkv_t + 2 * sDD, Dm, Dm,
                                                   sDD, 3 * sDD);
    conv_transpose_kernel<<<tDD, 256, 0, stream>>>(wo, wo_t, Dm, Dm, sDD, sDD);
    conv_transpose_kernel<<<dim3(4 * Dm / 32, Dm / 32, Ll), 256, 0, stream>>>(
        m1w, m1w_t, Dm, 4 * Dm, sD4, sD4);
    conv_transpose_kernel<<<dim3(Dm / 32, 4 * Dm / 32, Ll), 256, 0, stream>>>(
        m2w, m2w_t, 4 * Dm, Dm, sD4, sD4);
    const size_t nemb = (size_t)Vv * Dm;
    cast_bf16_kernel<<<(unsigned)((nemb + 255) / 256), 256, 0, stream>>>(
        tok_emb, embbf, nemb);

    // ---- embedding + emb LN (fp32 residual stream) ----
    gather_embed_kernel<<<Tt, 256, 0, stream>>>(input_ids, tok_emb, x);
    ln_kernel<float><<<Tt, 256, 0, stream>>>(x, emb_ln_w, emb_ln_b, x);

    const dim3 gDD(Dm / 128, Tt / 128);          // 6 x 32
    const dim3 gQKV(3 * Dm / 128, Tt / 128);     // 18 x 32
    const dim3 gD4(4 * Dm / 128, Tt / 128);      // 24 x 32

    for (int l = 0; l < Ll; ++l) {
        // attention block
        ln_kernel<__bf16><<<Tt, 256, 0, stream>>>(x, n1w + l * Dm,
                                                  n1b + l * Dm, h_bf);
        gemm_bf16_kernel<<<gQKV, 256, 0, stream>>>(
            h_bf, wqkv_t + (size_t)l * 3 * sDD, qkv, nullptr, nullptr, nullptr,
            Tt, 3 * Dm, Dm);
        rope_kernel<<<(Tt * Hh * 32) / 256, 256, 0, stream>>>(qkv);
        ttt_scan_kernel<<<Bb * Hh, 256, 0, stream>>>(qkv, o_bf);
        gemm_bf16_kernel<<<gDD, 256, 0, stream>>>(
            o_bf, wo_t + (size_t)l * sDD, x, nullptr, nullptr, x, Tt, Dm, Dm);

        // MLP block (gelu fused into GEMM epilogue, bf16 out)
        ln_kernel<__bf16><<<Tt, 256, 0, stream>>>(x, n2w + l * Dm,
                                                  n2b + l * Dm, h_bf);
        gemm_bf16_kernel<<<gD4, 256, 0, stream>>>(
            h_bf, m1w_t + (size_t)l * sD4, nullptr, g_bf,
            m1b + (size_t)l * 4 * Dm, nullptr, Tt, 4 * Dm, Dm);
        gemm_bf16_kernel<<<gDD, 256, 0, stream>>>(
            g_bf, m2w_t + (size_t)l * sD4, x, nullptr, m2b + (size_t)l * Dm, x,
            Tt, Dm, 4 * Dm);
    }

    // ---- final LN + logits (N = V, clamped/guarded edge tiles) ----
    ln_kernel<__bf16><<<Tt, 256, 0, stream>>>(x, onw, onb, h_bf);
    gemm_bf16_kernel<<<dim3((Vv + 127) / 128, Tt / 128), 256, 0, stream>>>(
        h_bf, embbf, (float*)d_out, nullptr, nullptr, nullptr, Tt, Vv, Dm);
}
